// SparseMHA_42949672960441
// MI455X (gfx1250) — matrix operand order repaired and checked by reading, not compiled, not run on hardware
//
#include <hip/hip_runtime.h>

// ---------------------------------------------------------------------------
// Sparse multi-head attention (GAT-style) for MI455X / gfx1250, wave32.
//   N nodes, HID=128, H=8 heads, D=16, E edges.
//   Q/K/V + output projections: bf16 WMMA (v_wmma_f32_16x16x32_bf16) with
//   LDS-staged A tiles (shared by the 8 waves of a block) and b128 B loads.
//   Edge phase: bf16 gathers from L2-resident Q/K/V, wave-per-edge.
// ---------------------------------------------------------------------------

typedef __attribute__((ext_vector_type(16))) __bf16 v16bf;
typedef __attribute__((ext_vector_type(8)))  __bf16 v8bf;
typedef __attribute__((ext_vector_type(8)))  float  v8f;
typedef __attribute__((ext_vector_type(4)))  __bf16 v4bf;

#define HID 128
#define NHEAD 8

// Monotone float <-> uint encoding so atomicMax(u32) == float max.
__device__ __forceinline__ unsigned enc_f32(float f) {
  unsigned u = __float_as_uint(f);
  return (u & 0x80000000u) ? ~u : (u | 0x80000000u);
}
__device__ __forceinline__ float dec_f32(unsigned u) {
  return __uint_as_float((u & 0x80000000u) ? (u & 0x7FFFFFFFu) : ~u);
}
#define ENC_NEG_INF 0x007FFFFFu  // enc(-inf)

// ---------------------------------------------------------------------------
// WMMA GEMM:  out[i,j] = (sum_k X[i,k] * W[j,k] + bias[j]) * scale
// Block = 8 waves = one 16-row strip of X across all 8 feature tiles.
// The 16x128 X tile is converted to bf16 in LDS once per block; each wave
// slices its A fragments out of LDS with 16-byte ds loads.
// A frag (16x32 bf16): lane<16 -> M=lane, K=0..15; lane>=16 -> K=16..31
// B frag (32x16 bf16): lane<16 -> K=0..15, col N=lane; lane>=16 -> K=16..31
// C/D (16x16 f32): lane<16 -> rows M=0..7 in c[0..7]; lane>=16 -> M=8..15
// ---------------------------------------------------------------------------
template <bool OUT_BF16>
__global__ __launch_bounds__(256) void gemm_bias_kernel(
    const float* __restrict__ X, const float* __restrict__ W,
    const float* __restrict__ bias, void* __restrict__ out,
    int n_rows, float scale) {
  __shared__ __bf16 tileA[16 * HID];  // 4 KB

  const int i0 = blockIdx.x * 16;     // node tile
  const int tj = threadIdx.x >> 5;    // feature tile (one per wave)
  const int lane = threadIdx.x & 31;
  const int half = lane >> 4;
  const int lm = lane & 15;

  // --- stage X tile -> LDS (bf16). 256 threads x 8 elements = 16x128. ---
  {
    const int idx = threadIdx.x * 8;      // element index in tile
    const int r = idx >> 7;               // tile row 0..15
    const int c0 = idx & 127;             // col, multiple of 8
    int srow = i0 + r;
    srow = (srow < n_rows) ? srow : (n_rows - 1);  // clamp: no branchy loads
    const float* src = X + (size_t)srow * HID + c0;
    const float4 f0 = *(const float4*)(src);
    const float4 f1 = *(const float4*)(src + 4);
    v8bf pk;
    pk[0] = (__bf16)f0.x; pk[1] = (__bf16)f0.y;
    pk[2] = (__bf16)f0.z; pk[3] = (__bf16)f0.w;
    pk[4] = (__bf16)f1.x; pk[5] = (__bf16)f1.y;
    pk[6] = (__bf16)f1.z; pk[7] = (__bf16)f1.w;
    *(v8bf*)(tileA + idx) = pk;
  }
  __syncthreads();

  const __bf16* arow_lds = tileA + lm * HID;
  const float* wrow = W + (size_t)(tj * 16 + lm) * HID;  // B col = W row

  v8f c = {};
#pragma unroll
  for (int kk = 0; kk < HID; kk += 32) {
    // A fragment: two 16-byte LDS loads (runs of 8 bf16).
    const v8bf r0 = *(const v8bf*)(arow_lds + kk + half * 16);
    const v8bf r1 = *(const v8bf*)(arow_lds + kk + half * 16 + 8);
    v16bf a;
#pragma unroll
    for (int i = 0; i < 8; ++i) { a[i] = r0[i]; a[i + 8] = r1[i]; }

    // B fragment: 16 contiguous floats of W -> 4 x global_load_b128 + cvt_pk.
    const float* wp = wrow + kk + half * 16;
    const float4 w0 = *(const float4*)(wp + 0);
    const float4 w1 = *(const float4*)(wp + 4);
    const float4 w2 = *(const float4*)(wp + 8);
    const float4 w3 = *(const float4*)(wp + 12);
    v16bf b;
    b[0]  = (__bf16)w0.x; b[1]  = (__bf16)w0.y; b[2]  = (__bf16)w0.z; b[3]  = (__bf16)w0.w;
    b[4]  = (__bf16)w1.x; b[5]  = (__bf16)w1.y; b[6]  = (__bf16)w1.z; b[7]  = (__bf16)w1.w;
    b[8]  = (__bf16)w2.x; b[9]  = (__bf16)w2.y; b[10] = (__bf16)w2.z; b[11] = (__bf16)w2.w;
    b[12] = (__bf16)w3.x; b[13] = (__bf16)w3.y; b[14] = (__bf16)w3.z; b[15] = (__bf16)w3.w;

    c = __builtin_amdgcn_wmma_f32_16x16x32_bf16(
        /*neg_a=*/false, a, /*neg_b=*/false, b,
        /*c_mod=*/(short)0, c, /*reuse_a=*/false, /*reuse_b=*/false);
  }

  const int feat = tj * 16 + lm;
  const float bv = bias[feat];
#pragma unroll
  for (int r = 0; r < 8; ++r) {
    const int node = i0 + half * 8 + r;
    if (node < n_rows) {
      const float val = (c[r] + bv) * scale;
      if (OUT_BF16)
        ((__bf16*)out)[(size_t)node * HID + feat] = (__bf16)val;
      else
        ((float*)out)[(size_t)node * HID + feat] = val;
    }
  }
}

// ---------------------------------------------------------------------------
// Init workspace: maxenc = enc(-inf), sums = 0, acc = 0.
// ---------------------------------------------------------------------------
__global__ void init_ws_kernel(unsigned* __restrict__ maxenc,
                               float* __restrict__ sumbuf,
                               float* __restrict__ acc, int n8, int n128) {
  const int i = blockIdx.x * blockDim.x + threadIdx.x;
  if (i < n8) { maxenc[i] = ENC_NEG_INF; sumbuf[i] = 0.0f; }
  if (i < n128) acc[i] = 0.0f;
}

// ---------------------------------------------------------------------------
// bsddmm: per-edge multi-head dot, wave per edge.
// Feature f = d*8 + h  ->  head = f & 7. Lane loads 4 contiguous bf16 (8 B):
// even lanes cover heads 0..3, odd lanes heads 4..7. Parity-preserving xor
// reduction (2,4,8,16) leaves head totals 0..3 in lane 0, 4..7 in lane 1.
// ---------------------------------------------------------------------------
__global__ __launch_bounds__(256) void edge_logits_kernel(
    const __bf16* __restrict__ Q, const __bf16* __restrict__ K,
    const int* __restrict__ row, const int* __restrict__ col,
    float* __restrict__ logits, unsigned* __restrict__ maxenc, int E) {
  const int e = blockIdx.x * (blockDim.x >> 5) + (threadIdx.x >> 5);
  if (e >= E) return;
  const int lane = threadIdx.x & 31;
  const int r = row[e];
  const int cc = col[e];

  const v4bf qv = *(const v4bf*)(Q + (size_t)r * HID + lane * 4);
  const v4bf kv = *(const v4bf*)(K + (size_t)cc * HID + lane * 4);
  float s[4];
#pragma unroll
  for (int i = 0; i < 4; ++i) s[i] = (float)qv[i] * (float)kv[i];

#pragma unroll
  for (int m = 2; m <= 16; m <<= 1) {
#pragma unroll
    for (int i = 0; i < 4; ++i) s[i] += __shfl_xor(s[i], m, 32);
  }

  if (lane < 2) {
#pragma unroll
    for (int i = 0; i < 4; ++i) {
      const int h = lane * 4 + i;
      const float lg = s[i];
      logits[(size_t)e * NHEAD + h] = lg;
      atomicMax(&maxenc[(size_t)r * NHEAD + h], enc_f32(lg));
    }
  }
}

// ---------------------------------------------------------------------------
// exp(logit - segmax) in-place + atomic segment-sum. Thread per (edge, head).
// ---------------------------------------------------------------------------
__global__ __launch_bounds__(256) void edge_exp_kernel(
    const int* __restrict__ row, float* __restrict__ logits,
    const unsigned* __restrict__ maxenc, float* __restrict__ sumbuf,
    int total) {
  const int idx = blockIdx.x * blockDim.x + threadIdx.x;
  if (idx >= total) return;
  const int e = idx >> 3;
  const int h = idx & 7;
  const int r = row[e];
  float m = dec_f32(maxenc[(size_t)r * NHEAD + h]);
  if (!(m > -3.0e38f)) m = 0.0f;  // guard empty segments (-inf)
  const float ex = __expf(logits[idx] - m);
  logits[idx] = ex;
  atomicAdd(&sumbuf[(size_t)r * NHEAD + h], ex);
}

// ---------------------------------------------------------------------------
// bspmm: acc[row, f] += (ex/sum) * V[col, f], wave per edge. Lanes 0..7 build
// the 8 per-head coefficients; each lane pulls the coeff for its 4 features
// via shuffle and scatters with f32 atomics.
// ---------------------------------------------------------------------------
__global__ __launch_bounds__(256) void edge_spmm_kernel(
    const __bf16* __restrict__ V, const int* __restrict__ row,
    const int* __restrict__ col, const float* __restrict__ ex,
    const float* __restrict__ sumbuf, float* __restrict__ acc, int E) {
  const int e = blockIdx.x * (blockDim.x >> 5) + (threadIdx.x >> 5);
  if (e >= E) return;
  const int lane = threadIdx.x & 31;
  const int r = row[e];
  const int cc = col[e];

  float coef = 0.0f;
  if (lane < NHEAD) {
    const float s = sumbuf[(size_t)r * NHEAD + lane];
    coef = ex[(size_t)e * NHEAD + lane] / (s == 0.0f ? 1.0f : s);
  }

  const v4bf vv = *(const v4bf*)(V + (size_t)cc * HID + lane * 4);
#pragma unroll
  for (int i = 0; i < 4; ++i) {
    const int h = (lane * 4 + i) & 7;
    const float cf = __shfl(coef, h, 32);
    atomicAdd(&acc[(size_t)r * HID + lane * 4 + i], cf * (float)vv[i]);
  }
}

// ---------------------------------------------------------------------------
static inline size_t align256(size_t x) { return (x + 255) & ~(size_t)255; }

extern "C" void kernel_launch(void* const* d_in, const int* in_sizes, int n_in,
                              void* d_out, int out_size, void* d_ws,
                              size_t ws_size, hipStream_t stream) {
  (void)n_in; (void)out_size; (void)ws_size;
  const float* h  = (const float*)d_in[0];
  const int*   row = (const int*)d_in[1];
  const int*   col = (const int*)d_in[2];
  const float* Wq = (const float*)d_in[3];
  const float* bq = (const float*)d_in[4];
  const float* Wk = (const float*)d_in[5];
  const float* bk = (const float*)d_in[6];
  const float* Wv = (const float*)d_in[7];
  const float* bv = (const float*)d_in[8];
  const float* Wo = (const float*)d_in[9];
  const float* bo = (const float*)d_in[10];
  float* out = (float*)d_out;

  const int N = in_sizes[0] / HID;
  const int E = in_sizes[1];
  const float SCALING = 0.25f;  // D^-0.5, D = 16

  // Workspace layout (bf16 Q/K/V keep the edge gathers L2-resident).
  char* ws = (char*)d_ws;
  size_t off = 0;
  __bf16* Qb = (__bf16*)(ws + off); off = align256(off + (size_t)N * HID * 2);
  __bf16* Kb = (__bf16*)(ws + off); off = align256(off + (size_t)N * HID * 2);
  __bf16* Vb = (__bf16*)(ws + off); off = align256(off + (size_t)N * HID * 2);
  float*  logits = (float*)(ws + off); off = align256(off + (size_t)E * NHEAD * 4);
  unsigned* maxenc = (unsigned*)(ws + off); off = align256(off + (size_t)N * NHEAD * 4);
  float*  sumbuf = (float*)(ws + off); off = align256(off + (size_t)N * NHEAD * 4);
  float*  acc = (float*)(ws + off); off = align256(off + (size_t)N * HID * 4);

  // 0) init reduction buffers + accumulator
  {
    const int n128 = N * HID, n8 = N * NHEAD;
    init_ws_kernel<<<(n128 + 255) / 256, 256, 0, stream>>>(maxenc, sumbuf, acc,
                                                           n8, n128);
  }

  // 1-3) Q/K/V projections, bf16 out (Q pre-scaled).
  const int gemm_blocks = (N + 15) / 16;
  gemm_bias_kernel<true><<<gemm_blocks, 256, 0, stream>>>(h, Wq, bq, Qb, N, SCALING);
  gemm_bias_kernel<true><<<gemm_blocks, 256, 0, stream>>>(h, Wk, bk, Kb, N, 1.0f);
  gemm_bias_kernel<true><<<gemm_blocks, 256, 0, stream>>>(h, Wv, bv, Vb, N, 1.0f);

  // 4) per-edge multi-head dot + segment max
  const int edge_blocks = (E + 7) / 8;  // 8 waves / block, wave per edge
  edge_logits_kernel<<<edge_blocks, 256, 0, stream>>>(Qb, Kb, row, col, logits,
                                                      maxenc, E);

  // 5) exp + segment sum
  edge_exp_kernel<<<(E * NHEAD + 255) / 256, 256, 0, stream>>>(row, logits,
                                                               maxenc, sumbuf,
                                                               E * NHEAD);

  // 6) normalized scatter-accumulate (bspmm)
  edge_spmm_kernel<<<edge_blocks, 256, 0, stream>>>(Vb, row, col, logits,
                                                    sumbuf, acc, E);

  // 7) output projection, f32 out
  gemm_bias_kernel<false><<<gemm_blocks, 256, 0, stream>>>(acc, Wo, bo, out, N,
                                                           1.0f);
}
